// NodeformerProcessor_28999619182971
// MI455X (gfx1250) — matrix operand order repaired and checked by reading, not compiled, not run on hardware
//
#include <hip/hip_runtime.h>
#include <hip/hip_bf16.h>
#include <math.h>

// ---------------- problem constants (from reference) ----------------
#define NN    50000          // nodes (== 3125 * 16, exact M-tiling)
#define DDIM  64             // head dim == in/out channels
#define NH    8              // heads
#define HDIM  512            // NH * DDIM
#define MF    30             // random features
#define MP    32             // padded random features (K multiple of 32)
#define ROWSQK (NN * NH)     // 400000 rows for per-head feature GEMMs

#define DN_SCALE   0.35355339059327373f   // 64^(-1/4)
#define RATIO_M    0.18257418583505536f   // 1/sqrt(30)
#define EPS_KF     1e-6f
#define EPS_NF     1e-5f
#define SLOPEF     0.01f

typedef __attribute__((ext_vector_type(16))) __bf16 v16bf;
typedef __attribute__((ext_vector_type(8)))  __bf16 v8bf;
typedef __attribute__((ext_vector_type(8)))  float  v8f;
typedef __attribute__((ext_vector_type(4)))  float  f4;

// ---------------- WMMA register-layout helpers (CDNA5, wave32) ----------------
// A (16-bit 16x32): lane row = l&15; element e: K = (e<8 ? e : 16+e-8) + 8*half
//   -> per lane, two contiguous runs of 8 (vectorizable).
__device__ __forceinline__ int ka_idx(int half, int e) {
  return ((e < 8) ? e : (8 + e)) + half * 8;   // e<8: e+8h ; e>=8: 16+(e-8)+8h
}
// B (16-bit 32x16): lane col = l&15; K = same order as A (two runs of 8).
__device__ __forceinline__ int kb_idx(int half, int e) {
  return ((e < 8) ? e : (8 + e)) + half * 8;
}

__device__ __forceinline__ void atomicMaxF(float* addr, float val) {
  if (val >= 0.f) atomicMax((int*)addr, __float_as_int(val));
  else            atomicMin((unsigned int*)addr, __float_as_uint(val));
}

// ---------------- small utility kernels ----------------
__global__ void fill_f32(float* __restrict__ p, float v, int n) {
  int i = blockIdx.x * blockDim.x + threadIdx.x;
  if (i < n) p[i] = v;
}

// W [K x C] row-major f32  ->  Wt [C x K] bf16 (B-operand friendly layout)
__global__ void cvt_wt(const float* __restrict__ W, __bf16* __restrict__ Wt, int K, int C) {
  int i = blockIdx.x * blockDim.x + threadIdx.x;
  if (i >= K * C) return;
  int c = i / K, k = i % K;
  Wt[i] = (__bf16)W[(size_t)k * C + c];
}

// P [30 x 64] -> Pt bf16 [32 x 64], rows m=30,31 zero (already [col x K] layout)
__global__ void cvt_ppad(const float* __restrict__ P, __bf16* __restrict__ Pt) {
  int i = blockIdx.x * blockDim.x + threadIdx.x;
  if (i >= MP * DDIM) return;
  int m = i / DDIM;
  Pt[i] = (m < MF) ? (__bf16)P[i] : (__bf16)0.f;
}

// ---------------- GEMM: out[rows x ncol] = (X*amul) @ W + bias ----------------
// X: [rows x 64] f32. Wt: bf16 [ncol x 64] (pre-transposed). one wave / 16x16 tile.
__global__ void gemm_xw(const float* __restrict__ X, float amul,
                        const __bf16* __restrict__ Wt, const float* __restrict__ bias,
                        float* __restrict__ out, int ncol) {
  const int rt = blockIdx.x, ct = blockIdx.y;
  const int l = threadIdx.x, lane = l & 15, half = l >> 4;
  const int row = rt * 16 + lane;
  const int col = ct * 16 + lane;
  const float*  Xr = X  + (size_t)row * DDIM;
  const __bf16* Wr = Wt + (size_t)col * DDIM;
  const float bv = bias ? bias[col] : 0.f;       // hoisted: one load, no per-store branch
  v8f acc = {};
#pragma unroll
  for (int ks = 0; ks < 2; ++ks) {               // K = 64 = 2 x 32
    const int o0 = ks * 32 + 8 * half;           // A run 0: K = o0..o0+7
    const int o1 = o0 + 16;                      // A run 1
    f4 x0 = *(const f4*)(Xr + o0);
    f4 x1 = *(const f4*)(Xr + o0 + 4);
    f4 x2 = *(const f4*)(Xr + o1);
    f4 x3 = *(const f4*)(Xr + o1 + 4);
    v8bf b0 = *(const v8bf*)(Wr + o0);
    v8bf b1 = *(const v8bf*)(Wr + o1);
    v16bf a, b;
#pragma unroll
    for (int e = 0; e < 4; ++e) {
      a[e]      = (__bf16)(x0[e] * amul);
      a[e + 4]  = (__bf16)(x1[e] * amul);
      a[e + 8]  = (__bf16)(x2[e] * amul);
      a[e + 12] = (__bf16)(x3[e] * amul);
    }
#pragma unroll
    for (int e = 0; e < 8; ++e) { b[e] = b0[e]; b[e + 8] = b1[e]; }
    acc = __builtin_amdgcn_wmma_f32_16x16x32_bf16(false, a, false, b, (short)0, acc, false, false);
  }
  float* orow = out + (size_t)(rt * 16 + 8 * half) * ncol + col;
#pragma unroll
  for (int g = 0; g < 8; ++g)
    orow[(size_t)g * ncol] = acc[g] + bv;
}

// ---------------- per-row diag + stabilizer ----------------
__global__ void row_stats(const float* __restrict__ qk, const float* __restrict__ dd,
                          float* __restrict__ diag, float* __restrict__ stab, int is_query) {
  int row = blockIdx.x * blockDim.x + threadIdx.x;
  if (row >= ROWSQK) return;
  float s2 = 0.f;
  for (int i = 0; i < DDIM; i += 4) {
    f4 q = *(const f4*)(qk + (size_t)row * DDIM + i);
    s2 += q.x * q.x + q.y * q.y + q.z * q.z + q.w * q.w;
  }
  diag[row] = s2 * (0.5f * DN_SCALE * DN_SCALE);
  float mx = -3.0e38f;
  for (int m = 0; m < MF; ++m) mx = fmaxf(mx, dd[(size_t)row * MP + m]);
  if (is_query) stab[row] = mx;
  else          atomicMaxF(&stab[row % NH], mx);   // row = n*NH + h
}

// dd -> phi = ratio*(exp(dd - diag - stab) + eps); padded m -> 0 (in place)
__global__ void phi_exp(float* __restrict__ dd, const float* __restrict__ diag,
                        const float* __restrict__ stab, int is_query) {
  size_t idx = (size_t)blockIdx.x * blockDim.x + threadIdx.x;
  if (idx >= (size_t)ROWSQK * MP) return;
  int m = (int)(idx & 31); size_t row = idx >> 5;
  float o = 0.f;
  if (m < MF) {
    float s = is_query ? stab[row] : stab[(int)(row % NH)];
    o = RATIO_M * (__expf(dd[idx] - diag[row] - s) + EPS_KF);
  }
  dd[idx] = o;
}

// ---------------- kvs[h,m,d] = sum_n kp[n,h,m] * v[n,h,d] (WMMA, K over n) ----------------
__global__ void kvs_wmma(const float* __restrict__ kp, const float* __restrict__ v,
                         float* __restrict__ kvs) {
  const int chunk = blockIdx.x, h = blockIdx.y;
  const int l = threadIdx.x, lane = l & 15, half = l >> 4;
  const int total_ks = (NN + 31) / 32;     // 1563
  const int ks0 = chunk * 16;
  const int ksn = min(16, total_ks - ks0);
  v8f acc[2][4] = {};
  for (int ks = 0; ks < ksn; ++ks) {
    int nb = (ks0 + ks) * 32;
    v16bf a[2], b[4];
#pragma unroll
    for (int e = 0; e < 16; ++e) {
      int na = nb + ka_idx(half, e);
      bool va = na < NN;
      size_t ra = va ? ((size_t)na * NH + h) * MP : 0;
#pragma unroll
      for (int mt = 0; mt < 2; ++mt)
        a[mt][e] = va ? (__bf16)kp[ra + mt * 16 + lane] : (__bf16)0.f;
      int nk = nb + kb_idx(half, e);
      bool vb = nk < NN;
      size_t rb = vb ? ((size_t)nk * NH + h) * DDIM : 0;
#pragma unroll
      for (int dt = 0; dt < 4; ++dt)
        b[dt][e] = vb ? (__bf16)v[rb + dt * 16 + lane] : (__bf16)0.f;
    }
#pragma unroll
    for (int mt = 0; mt < 2; ++mt)
#pragma unroll
      for (int dt = 0; dt < 4; ++dt)
        acc[mt][dt] = __builtin_amdgcn_wmma_f32_16x16x32_bf16(
            false, a[mt], false, b[dt], (short)0, acc[mt][dt], false, false);
  }
#pragma unroll
  for (int mt = 0; mt < 2; ++mt)
    for (int dt = 0; dt < 4; ++dt)
#pragma unroll
      for (int g = 0; g < 8; ++g) {
        int m = mt * 16 + g + 8 * half;
        int d = dt * 16 + lane;
        atomicAdd(&kvs[((size_t)h * MP + m) * DDIM + d], acc[mt][dt][g]);
      }
}

// ksum[h*32+m] = sum_n kp[n, h*32+m]   (kp viewed as [N x 256])
__global__ void ksum_reduce(const float* __restrict__ kp, float* __restrict__ ksum) {
  __shared__ float sm[256];
  int col = blockIdx.x;     // 0..255
  float s = 0.f;
  for (int n = threadIdx.x; n < NN; n += 256)
    s += kp[(size_t)n * (NH * MP) + col];
  sm[threadIdx.x] = s; __syncthreads();
  for (int st = 128; st > 0; st >>= 1) {
    if ((int)threadIdx.x < st) sm[threadIdx.x] += sm[threadIdx.x + st];
    __syncthreads();
  }
  if (threadIdx.x == 0) ksum[col] = sm[0];
}

// den[row] = qp[row,:] . ksum[h,:]
__global__ void den_kernel(const float* __restrict__ qp, const float* __restrict__ ksum,
                           float* __restrict__ den) {
  int row = blockIdx.x * blockDim.x + threadIdx.x;
  if (row >= ROWSQK) return;
  int h = row % NH;
  float s = 0.f;
  for (int m = 0; m < MP; m += 4) {
    f4 a = *(const f4*)(qp + (size_t)row * MP + m);
    f4 k = *(const f4*)(ksum + h * MP + m);
    s += a.x * k.x + a.y * k.y + a.z * k.z + a.w * k.w;
  }
  den[row] = s;
}

// ---------------- num[n,h,d] = qp[n,h,:] @ kvs[h,:,:]  (WMMA, K=32) ----------------
__global__ void num_wmma(const float* __restrict__ qp, const float* __restrict__ kvs,
                         float* __restrict__ num) {
  const int rt = blockIdx.x, dt = blockIdx.y, h = blockIdx.z;
  const int l = threadIdx.x, lane = l & 15, half = l >> 4;
  const int n = rt * 16 + lane;
  const float* Qr = qp + ((size_t)n * NH + h) * MP;
  const int o0 = 8 * half, o1 = o0 + 16;
  f4 x0 = *(const f4*)(Qr + o0);
  f4 x1 = *(const f4*)(Qr + o0 + 4);
  f4 x2 = *(const f4*)(Qr + o1);
  f4 x3 = *(const f4*)(Qr + o1 + 4);
  v16bf a, b;
#pragma unroll
  for (int e = 0; e < 4; ++e) {
    a[e] = (__bf16)x0[e]; a[e + 4] = (__bf16)x1[e];
    a[e + 8] = (__bf16)x2[e]; a[e + 12] = (__bf16)x3[e];
  }
#pragma unroll
  for (int e = 0; e < 16; ++e)
    b[e] = (__bf16)kvs[((size_t)h * MP + kb_idx(half, e)) * DDIM + dt * 16 + lane];
  v8f acc = {};
  acc = __builtin_amdgcn_wmma_f32_16x16x32_bf16(false, a, false, b, (short)0, acc, false, false);
#pragma unroll
  for (int g = 0; g < 8; ++g) {
    int nr = rt * 16 + g + 8 * half;
    num[((size_t)nr * NH + h) * DDIM + dt * 16 + lane] = acc[g];
  }
}

// ---------------- y = leaky?((num/den) @ Wo + bo); accumulate col sums ----------------
// Wot: bf16 [64 x 512] (pre-transposed Wo)
__global__ void gemm_out(const float* __restrict__ num, const float* __restrict__ den,
                         const __bf16* __restrict__ Wot, const float* __restrict__ bo,
                         float* __restrict__ y, float* __restrict__ colsum,
                         float* __restrict__ colsumsq, int do_act) {
  const int rt = blockIdx.x, dt = blockIdx.y;
  const int l = threadIdx.x, lane = l & 15, half = l >> 4;
  const int n = rt * 16 + lane;         // fixed A-row per lane
  const int col = dt * 16 + lane;
  const float*  Nr = num + (size_t)n * HDIM;    // [N][H*D] view
  const __bf16* Wr = Wot + (size_t)col * HDIM;
  const float bv = bo[col];
  float rden[NH];
#pragma unroll
  for (int h = 0; h < NH; ++h) rden[h] = 1.f / den[(size_t)n * NH + h];
  v8f acc = {};
#pragma unroll
  for (int ks = 0; ks < 16; ++ks) {     // K = 512 = 16 x 32
    const int o0 = ks * 32 + 8 * half;  // 8-aligned -> stays inside one head
    const int o1 = o0 + 16;
    f4 x0 = *(const f4*)(Nr + o0);
    f4 x1 = *(const f4*)(Nr + o0 + 4);
    f4 x2 = *(const f4*)(Nr + o1);
    f4 x3 = *(const f4*)(Nr + o1 + 4);
    float r0 = rden[o0 >> 6], r1 = rden[o1 >> 6];
    v8bf b0 = *(const v8bf*)(Wr + o0);
    v8bf b1 = *(const v8bf*)(Wr + o1);
    v16bf a, b;
#pragma unroll
    for (int e = 0; e < 4; ++e) {
      a[e]      = (__bf16)(x0[e] * r0);
      a[e + 4]  = (__bf16)(x1[e] * r0);
      a[e + 8]  = (__bf16)(x2[e] * r1);
      a[e + 12] = (__bf16)(x3[e] * r1);
    }
#pragma unroll
    for (int e = 0; e < 8; ++e) { b[e] = b0[e]; b[e + 8] = b1[e]; }
    acc = __builtin_amdgcn_wmma_f32_16x16x32_bf16(false, a, false, b, (short)0, acc, false, false);
  }
  float s1 = 0.f, s2 = 0.f;
  float* yrow = y + (size_t)(rt * 16 + 8 * half) * DDIM + col;
#pragma unroll
  for (int g = 0; g < 8; ++g) {
    float val = acc[g] + bv;
    if (do_act) val = (val >= 0.f) ? val : SLOPEF * val;
    yrow[(size_t)g * DDIM] = val;
    s1 += val; s2 += val * val;
  }
  atomicAdd(&colsum[col], s1);
  atomicAdd(&colsumsq[col], s2);
}

// GraphNorm: scale = gamma*rsqrt(var+eps); shift = beta - scale*alpha*mu
__global__ void norm_finalize(const float* __restrict__ cs, const float* __restrict__ cs2,
                              const float* __restrict__ gamma, const float* __restrict__ beta,
                              const float* __restrict__ alpha,
                              float* __restrict__ scale, float* __restrict__ shift) {
  int c = threadIdx.x;
  if (c >= DDIM) return;
  const float invN = 1.f / (float)NN;
  float mu  = cs[c] * invN;
  float ex2 = cs2[c] * invN;
  float a   = alpha[c];
  float var = ex2 - (2.f * a - a * a) * mu * mu;   // E[(x-a*mu)^2]
  float s = gamma[c] * rsqrtf(var + EPS_NF);
  scale[c] = s;
  shift[c] = beta[c] - s * a * mu;
}

__global__ void norm_apply(const float* __restrict__ y, const float* __restrict__ scale,
                           const float* __restrict__ shift, float* __restrict__ out) {
  size_t i = (size_t)blockIdx.x * blockDim.x + threadIdx.x;
  if (i >= (size_t)NN * DDIM) return;
  int c = (int)(i & 63);
  out[i] = scale[c] * y[i] + shift[c];
}

// ---------------- host launch ----------------
extern "C" void kernel_launch(void* const* d_in, const int* in_sizes, int n_in,
                              void* d_out, int out_size, void* d_ws, size_t ws_size,
                              hipStream_t stream) {
  (void)in_sizes; (void)n_in; (void)out_size; (void)ws_size;
  const float* patch = (const float*)d_in[0];
  const float* Wq = (const float*)d_in[3];
  const float* Wk = (const float*)d_in[4];
  const float* Wv = (const float*)d_in[5];
  const float* Wo = (const float*)d_in[6];
  const float* bq = (const float*)d_in[7];
  const float* bk = (const float*)d_in[8];
  const float* bv = (const float*)d_in[9];
  const float* bo = (const float*)d_in[10];
  const float* P  = (const float*)d_in[11];
  const float* gamma = (const float*)d_in[12];
  const float* beta  = (const float*)d_in[13];
  const float* alpha = (const float*)d_in[14];

  char* ws = (char*)d_ws;
  size_t off = 0;
  auto alloc = [&](size_t bytes) -> void* {
    void* p = (void*)(ws + off);
    off += (bytes + 255) & ~(size_t)255;
    return p;
  };
  float* qf    = (float*)alloc((size_t)NN * HDIM * 4);     // also num buffer
  float* kf    = (float*)alloc((size_t)NN * HDIM * 4);
  float* vf    = (float*)alloc((size_t)NN * HDIM * 4);
  float* qpf   = (float*)alloc((size_t)ROWSQK * MP * 4);
  float* kpf   = (float*)alloc((size_t)ROWSQK * MP * 4);   // also y buffer
  float* xn    = (float*)alloc((size_t)NN * DDIM * 4);     // normalized activations
  float* denf  = (float*)alloc((size_t)ROWSQK * 4);
  float* diagf = (float*)alloc((size_t)ROWSQK * 4);
  float* stabq = (float*)alloc((size_t)ROWSQK * 4);
  float* kstab = (float*)alloc(NH * 4);
  float* ksumf = (float*)alloc(NH * MP * 4);
  float* kvsf  = (float*)alloc((size_t)NH * MP * DDIM * 4);
  float* csum  = (float*)alloc(DDIM * 4);
  float* csum2 = (float*)alloc(DDIM * 4);
  float* scl   = (float*)alloc(DDIM * 4);
  float* shf   = (float*)alloc(DDIM * 4);
  __bf16* wqt = (__bf16*)alloc((size_t)HDIM * DDIM * 2);   // [512 x 64]
  __bf16* wkt = (__bf16*)alloc((size_t)HDIM * DDIM * 2);
  __bf16* wvt = (__bf16*)alloc((size_t)HDIM * DDIM * 2);
  __bf16* wot = (__bf16*)alloc((size_t)DDIM * HDIM * 2);   // [64 x 512]
  __bf16* ptb = (__bf16*)alloc((size_t)MP * DDIM * 2);     // [32 x 64]

  const int NT = NN / 16;                            // 3125
  const int KCHUNKS = ((NN + 31) / 32 + 15) / 16;    // 98
  const int WN = DDIM * HDIM;                        // 32768

  const float* x = patch;

  for (int L = 0; L < 3; ++L) {
    // weight conversions (f32 -> bf16, transposed into [col x K])
    cvt_wt<<<(WN + 255) / 256, 256, 0, stream>>>(Wq + (size_t)L * WN, wqt, DDIM, HDIM);
    cvt_wt<<<(WN + 255) / 256, 256, 0, stream>>>(Wk + (size_t)L * WN, wkt, DDIM, HDIM);
    cvt_wt<<<(WN + 255) / 256, 256, 0, stream>>>(Wv + (size_t)L * WN, wvt, DDIM, HDIM);
    cvt_wt<<<(WN + 255) / 256, 256, 0, stream>>>(Wo + (size_t)L * WN, wot, HDIM, DDIM);
    cvt_ppad<<<(MP * DDIM + 255) / 256, 256, 0, stream>>>(P + (size_t)L * MF * DDIM, ptb);

    // QKV projections
    gemm_xw<<<dim3(NT, HDIM / 16), 32, 0, stream>>>(x, 1.f, wqt, bq + L * HDIM, qf, HDIM);
    gemm_xw<<<dim3(NT, HDIM / 16), 32, 0, stream>>>(x, 1.f, wkt, bk + L * HDIM, kf, HDIM);
    gemm_xw<<<dim3(NT, HDIM / 16), 32, 0, stream>>>(x, 1.f, wvt, bv + L * HDIM, vf, HDIM);

    // query features
    gemm_xw<<<dim3(ROWSQK / 16, MP / 16), 32, 0, stream>>>(qf, DN_SCALE, ptb, nullptr, qpf, MP);
    row_stats<<<(ROWSQK + 255) / 256, 256, 0, stream>>>(qf, qpf, diagf, stabq, 1);
    phi_exp<<<(int)(((size_t)ROWSQK * MP + 255) / 256), 256, 0, stream>>>(qpf, diagf, stabq, 1);

    // key features (kpf overwrites y buffer — y already consumed into xn)
    gemm_xw<<<dim3(ROWSQK / 16, MP / 16), 32, 0, stream>>>(kf, DN_SCALE, ptb, nullptr, kpf, MP);
    fill_f32<<<1, 32, 0, stream>>>(kstab, -3.0e38f, NH);
    row_stats<<<(ROWSQK + 255) / 256, 256, 0, stream>>>(kf, kpf, diagf, kstab, 0);
    phi_exp<<<(int)(((size_t)ROWSQK * MP + 255) / 256), 256, 0, stream>>>(kpf, diagf, kstab, 0);

    // linear attention
    fill_f32<<<(NH * MP * DDIM + 255) / 256, 256, 0, stream>>>(kvsf, 0.f, NH * MP * DDIM);
    fill_f32<<<1, 64, 0, stream>>>(csum, 0.f, DDIM);
    fill_f32<<<1, 64, 0, stream>>>(csum2, 0.f, DDIM);
    kvs_wmma<<<dim3(KCHUNKS, NH), 32, 0, stream>>>(kpf, vf, kvsf);
    ksum_reduce<<<NH * MP, 256, 0, stream>>>(kpf, ksumf);
    den_kernel<<<(ROWSQK + 255) / 256, 256, 0, stream>>>(qpf, ksumf, denf);
    num_wmma<<<dim3(NT, DDIM / 16, NH), 32, 0, stream>>>(qpf, kvsf, qf);  // num aliases qf

    // output projection + activation + column stats (y aliases kpf)
    gemm_out<<<dim3(NT, DDIM / 16), 32, 0, stream>>>(qf, denf, wot, bo + L * DDIM,
                                                     kpf, csum, csum2, (L < 2) ? 1 : 0);
    norm_finalize<<<1, 64, 0, stream>>>(csum, csum2, gamma + L * DDIM, beta + L * DDIM,
                                        alpha + L * DDIM, scl, shf);
    if (L < 2) {
      norm_apply<<<(int)(((size_t)NN * DDIM + 255) / 256), 256, 0, stream>>>(kpf, scl, shf, xn);
      x = xn;
    }
  }

  // final GraphNorm applied into the output
  norm_apply<<<(int)(((size_t)NN * DDIM + 255) / 256), 256, 0, stream>>>(kpf, scl, shf, (float*)d_out);
}
